// SpectralChannelMixing_86406152061127
// MI455X (gfx1250) — matrix operand order repaired and checked by reading, not compiled, not run on hardware
//
#include <hip/hip_runtime.h>
#include <hip/hip_bf16.h>
#include <math.h>

// ---------------------------------------------------------------------------
// SpectralChannelMixing for MI455X (gfx1250, wave32, WMMA)
//   out = GELU_exact(x + blockdiag_matmul(x, wc))
//   x : [8, 512, 128, 128] f32   wc : [4, 128, 128] f32   out: same as x
// 32 GEMMs of M=128 (c_out) x K=128 (c_in) x N=16384 (pixels), f16 WMMA,
// f32 accumulate. Memory-bound (~537 MB @ 23.3 TB/s ~= 23 us floor).
//
// v3: single pixel-group per workgroup (no outer loop). A multi-iteration
// loop tempted the compiler to cache all A-fragments in VGPRs (~130 regs),
// exhausting the register file and serializing B-fragment loads into
// load -> s_wait_loadcnt 0 -> cvt chains. With one group per WG the live set
// is small, B loads batch into clauses, and occupancy rises. Wc re-staging
// per WG is L2-resident traffic only (wc = 256 KB total).
// Output uses non-temporal stores (write-once data; keep L2 for x/wc).
// ---------------------------------------------------------------------------

typedef _Float16 v16h __attribute__((ext_vector_type(16)));
typedef _Float16 h8   __attribute__((ext_vector_type(8)));
typedef _Float16 h2   __attribute__((ext_vector_type(2)));
typedef float    v8f  __attribute__((ext_vector_type(8)));

#define CBLK   4
#define CDIM   128
#define HWPIX  (128 * 128)   // pixels per batch image
#define PITCH  136           // LDS row pitch in halves (128 + 8 pad -> 4-bank stride, conflict free)

// Branch-free exact-GELU: 0.5*y*(1+erf(y/sqrt(2))), erf via A&S 7.1.26
// (|err| < 1.5e-7). rcp/exp2 are TRANS32 ops that co-execute with VALU.
__device__ __forceinline__ float gelu_erf(float y) {
    const float z  = y * 0.70710678118654752f;
    const float az = __builtin_fabsf(z);
    const float t  = __builtin_amdgcn_rcpf(__builtin_fmaf(0.3275911f, az, 1.0f));
    float p = 1.061405429f;
    p = __builtin_fmaf(p, t, -1.453152027f);
    p = __builtin_fmaf(p, t,  1.421413741f);
    p = __builtin_fmaf(p, t, -0.284496736f);
    p = __builtin_fmaf(p, t,  0.254829592f);
    p *= t;
    const float e = __builtin_amdgcn_exp2f(-az * az * 1.44269504088896340f);
    const float erf_abs = __builtin_fmaf(-p, e, 1.0f);
    const float erfv = __builtin_copysignf(erf_abs, z);
    return 0.5f * y * (1.0f + erfv);
}

__global__ __launch_bounds__(256)
void spectral_mix_wmma_kernel(const float* __restrict__ x,
                              const float* __restrict__ wc,
                              float* __restrict__ out)
{
    // Wc[j]^T staged as ws[c_out][c_in] in f16 (A-matrix source)
    __shared__ __align__(16) _Float16 ws[CDIM * PITCH];

    const int b    = blockIdx.y >> 2;   // batch
    const int j    = blockIdx.y & 3;    // channel block
    const int t    = threadIdx.x;
    const int lane = t & 31;
    const int wid  = t >> 5;            // wave id 0..7

    // ---- Stage Wc[j] (stored [c_in][c_out]) transposed into LDS as f16 ----
    {
        const float* wj  = wc + (long)j * CDIM * CDIM;
        const int co     = t & 127;           // contiguous in global -> coalesced
        const int ci0    = (t >> 7) * 64;     // two threads per c_out column
        for (int ci = ci0; ci < ci0 + 64; ci += 2) {
            float w0 = wj[(long)ci       * CDIM + co];
            float w1 = wj[(long)(ci + 1) * CDIM + co];
            h2 pk;
            pk.x = (_Float16)w0;
            pk.y = (_Float16)w1;
            *(h2*)&ws[co * PITCH + ci] = pk;  // ds_store_b32, 4-bank lane stride
        }
    }
    __syncthreads();

    const float* xb = x   + ((long)b * (CBLK * CDIM) + (long)j * CDIM) * HWPIX;
    float*       ob = out + ((long)b * (CBLK * CDIM) + (long)j * CDIM) * HWPIX;

    const int lhalf = lane >> 4;   // 0: lanes 0-15, 1: lanes 16-31 (branchless)
    const int l15   = lane & 15;

    // One 128-pixel group per workgroup; this wave owns 16 pixels.
    const int p = blockIdx.x * 128 + wid * 16 + l15;   // this lane's pixel column

    // ---- B fragments: X[k, p] for k=0..127, f32 -> f16, kept in regs ----
    // B 32x16 layout used here (contiguous k, same as A): lanes 0-15 hold K=kk+0..15 (N = lane),
    //                            lanes 16-31 hold K=kk+16..31.
    v16h bf[4];
#pragma unroll
    for (int c = 0; c < 4; ++c) {
        const int koff  = c * 32 + lhalf * 16;
        const float* xc = xb + (long)koff * HWPIX + p;
#pragma unroll
        for (int i = 0; i < 16; ++i)
            bf[c][i] = (_Float16)xc[(long)i * HWPIX];   // coalesced per half-wave
    }

    // ---- 8 M-tiles of 16 c_out, 4 WMMAs each (K=128) ----
#pragma unroll
    for (int mt = 0; mt < 8; ++mt) {
        v8f acc = (v8f){0.f, 0.f, 0.f, 0.f, 0.f, 0.f, 0.f, 0.f};
        const int row = mt * 16 + l15;   // A row = c_out (same for lane, lane+16)
#pragma unroll
        for (int c = 0; c < 4; ++c) {
            // A 16x32 layout (matches B): lanes 0-15: K = kk+0..15
            //                             lanes 16-31: K = kk+16..31
            const int k0 = c * 32 + lhalf * 16;
            h8 g0 = *(const h8*)&ws[row * PITCH + k0];        // ds_read_b128
            h8 g1 = *(const h8*)&ws[row * PITCH + k0 + 8];    // ds_read_b128
            v16h af = __builtin_shufflevector(
                g0, g1, 0, 1, 2, 3, 4, 5, 6, 7, 8, 9, 10, 11, 12, 13, 14, 15);
            acc = __builtin_amdgcn_wmma_f32_16x16x32_f16(
                /*neg_a=*/false, af, /*neg_b=*/false, bf[c],
                /*c_mod=*/(short)0, acc, /*reuse_a=*/false, /*reuse_b=*/false);
        }
        // ---- Epilogue: residual + exact (erf) GELU, coalesced NT stores ----
        // D layout: VGPR r -> M = mt*16 + r (lanes 0-15) / +8 (lanes 16-31)
#pragma unroll
        for (int r = 0; r < 8; ++r) {
            const int  co  = mt * 16 + lhalf * 8 + r;
            const long off = (long)co * HWPIX + p;
            const float y  = xb[off] + acc[r];     // residual (tile hot in cache)
            __builtin_nontemporal_store(gelu_erf(y), &ob[off]);
        }
    }
}

extern "C" void kernel_launch(void* const* d_in, const int* in_sizes, int n_in,
                              void* d_out, int out_size, void* d_ws, size_t ws_size,
                              hipStream_t stream) {
    const float* x  = (const float*)d_in[0];   // [8, 512, 128, 128] f32
    const float* wc = (const float*)d_in[1];   // [4, 128, 128] f32
    float* out      = (float*)d_out;           // [8, 512, 128, 128] f32
    (void)in_sizes; (void)n_in; (void)out_size; (void)d_ws; (void)ws_size;

    dim3 grid(HWPIX / 128, 8 * CBLK, 1);   // 128 pixel groups x 32 (b, j) pairs
    dim3 block(256, 1, 1);                 // 8 wave32 waves
    spectral_mix_wmma_kernel<<<grid, block, 0, stream>>>(x, wc, out);
}
